// LSTM_80831284510999
// MI455X (gfx1250) — matrix operand order repaired and checked by reading, not compiled, not run on hardware
//
#include <hip/hip_runtime.h>
#include <hip/hip_bf16.h>

typedef float v2f  __attribute__((ext_vector_type(2)));
typedef float v4f  __attribute__((ext_vector_type(4)));
typedef float v8f  __attribute__((ext_vector_type(8)));
typedef __bf16 bf16_t;
typedef bf16_t v8bf  __attribute__((ext_vector_type(8)));
typedef bf16_t v16bf __attribute__((ext_vector_type(16)));
typedef unsigned short v4us __attribute__((ext_vector_type(4)));

#define BB 4096
#define EE 1024
#define HH 1024
#define CC 1024
#define DD 3072
#define SWA 40   // LDS row stride in bf16 elements (80B, 16B-aligned rows)

#define WMMA_BF16(a, b, c) \
    __builtin_amdgcn_wmma_f32_16x16x32_bf16(false, (a), false, (b), (short)0, (c), false, false)

__device__ __forceinline__ float sigmoidf_(float x) {
    return 1.0f / (1.0f + __expf(-x));
}

// float -> bf16 (RNE), as raw u16 bits
__device__ __forceinline__ unsigned short f2bf_rne(float f) {
    unsigned int u = __float_as_uint(f);
    u += 0x7fffu + ((u >> 16) & 1u);
    return (unsigned short)(u >> 16);
}
// split x ~= hi + lo (both bf16); residual ~2^-16 relative
__device__ __forceinline__ void splitf(float x, unsigned short& h, unsigned short& l) {
    const unsigned short hs = f2bf_rne(x);
    const float hf = __uint_as_float(((unsigned int)hs) << 16);
    h = hs;
    l = f2bf_rne(x - hf);
}

// Async memory -> LDS copy, 16B per lane, tracked by ASYNCcnt.
// LDS byte offset = low 32 bits of the generic pointer (flat LDS aperture).
__device__ __forceinline__ void async_load_b128(void* lds_dst, const void* gsrc) {
    const unsigned lds = (unsigned)(unsigned long long)lds_dst;
    asm volatile("global_load_async_to_lds_b128 %0, %1, off"
                 :: "v"(lds), "v"((unsigned long long)gsrc)
                 : "memory");
}
__device__ __forceinline__ void wait_async0() {
    asm volatile("s_wait_asynccnt 0x0" ::: "memory");
}

// A fragment (16x32 bf16): lane-half g holds K {8g..8g+7} U {16+8g..16+8g+7}
__device__ __forceinline__ v16bf frag_a(const unsigned short* rowp, int g) {
    const v8bf p0 = *(const v8bf*)(rowp + g * 8);
    const v8bf p1 = *(const v8bf*)(rowp + 16 + g * 8);
    return __builtin_shufflevector(p0, p1, 0, 1, 2, 3, 4, 5, 6, 7, 8, 9, 10, 11, 12, 13, 14, 15);
}
// B fragment (32x16 bf16): lane-half g holds K {8g..8g+7} U {16+8g..16+8g+7}
__device__ __forceinline__ v16bf frag_b(const unsigned short* rowp, int g) {
    const v8bf p0 = *(const v8bf*)(rowp + g * 8);
    const v8bf p1 = *(const v8bf*)(rowp + 16 + g * 8);
    return __builtin_shufflevector(p0, p1, 0, 1, 2, 3, 4, 5, 6, 7, 8, 9, 10, 11, 12, 13, 14, 15);
}

// ---------------------------------------------------------------------------
// Weight decomposition: fp32 -> (hi, lo) bf16 planes. Streaming, one pass.
// ---------------------------------------------------------------------------
__global__ __launch_bounds__(256)
void decomp_kernel(const float* __restrict__ src, unsigned short* __restrict__ hi,
                   unsigned short* __restrict__ lo, int n4)
{
    const int i = blockIdx.x * 256 + threadIdx.x;
    if (i >= n4) return;
    const v4f v = *(const v4f*)(src + (size_t)i * 4);
    v4us h, l;
    #pragma unroll
    for (int j = 0; j < 4; ++j) {
        unsigned short hj, lj;
        splitf(v[j], hj, lj);
        h[j] = hj; l[j] = lj;
    }
    *(v4us*)(hi + (size_t)i * 4) = h;
    *(v4us*)(lo + (size_t)i * 4) = l;
}

// ---------------------------------------------------------------------------
// Kernel 1: 4 gate GEMMs (split-bf16) + LSTM elementwise -> hidden, cell
// Block tile 64(batch) x 64(h), Kt=32. 8 waves = 2(M-slab of 32) x 4(N of 16).
// Weight tiles staged via global_load_async_to_lds_b128 (ASYNCcnt).
// ---------------------------------------------------------------------------
__global__ __launch_bounds__(256)
void lstm_gates_bf16x2(const float* __restrict__ emb,
                       const float* __restrict__ hprev,
                       const float* __restrict__ cprev,
                       const float* __restrict__ ctx,
                       const unsigned short* __restrict__ WiH, const unsigned short* __restrict__ WiL,
                       const unsigned short* __restrict__ WfH, const unsigned short* __restrict__ WfL,
                       const unsigned short* __restrict__ WoH, const unsigned short* __restrict__ WoL,
                       const unsigned short* __restrict__ WgH, const unsigned short* __restrict__ WgL,
                       const float* __restrict__ pbi, const float* __restrict__ pbf,
                       const float* __restrict__ pbo, const float* __restrict__ pbg,
                       float* __restrict__ outHid, float* __restrict__ outCell)
{
    __shared__ __align__(16) unsigned short sA[2][64][SWA];     // [plane][batch row][K]
    __shared__ __align__(16) unsigned short sW[4][2][64][SWA];  // [gate][plane][h row][K]

    const int tid  = threadIdx.x;
    const int lane = tid & 31;
    const int wave = tid >> 5;
    const int nw   = wave & 3;         // N tile (16 h)
    const int mw   = wave >> 2;        // M slab (32 batch rows)
    const int row0 = blockIdx.x * 64;
    const int h0   = blockIdx.y * 64;
    const int fr   = lane & 15;
    const int g16  = lane >> 4;

    const unsigned short* WH[4] = {WiH, WfH, WoH, WgH};
    const unsigned short* WL[4] = {WiL, WfL, WoL, WgL};

    v8f acc[4][2] = {};

    for (int kt = 0; kt < DD; kt += 32) {
        // pick concat segment (1024 % 32 == 0: tiles never straddle)
        const float* asrc; int kl;
        if (kt < EE)           { asrc = emb;   kl = kt; }
        else if (kt < EE + HH) { asrc = hprev; kl = kt - EE; }
        else                   { asrc = ctx;   kl = kt - EE - HH; }

        // Issue async copies of all 8 weight plane tiles first (no VGPR traffic)
        {
            const int r = tid >> 2, c = (tid & 3) << 3;
            #pragma unroll
            for (int gpl = 0; gpl < 8; ++gpl) {
                const int g = gpl >> 1, p = gpl & 1;
                const unsigned short* src =
                    (p ? WL[g] : WH[g]) + (size_t)(h0 + r) * DD + kt + c;
                async_load_b128(&sW[g][p][r][c], src);
            }
        }
        // A tile: issue both global loads, then split (overlaps with async DMA)
        {
            v4f va[2];
            #pragma unroll
            for (int i = 0; i < 2; ++i) {
                const int idx = tid + i * 256;
                const int r = idx >> 3, c = (idx & 7) << 2;
                const float* gp = asrc + (size_t)(row0 + r) * 1024 + kl + c;
                if (kt + 32 < DD) __builtin_prefetch(gp + 32, 0, 1);
                va[i] = *(const v4f*)gp;
            }
            #pragma unroll
            for (int i = 0; i < 2; ++i) {
                const int idx = tid + i * 256;
                const int r = idx >> 3, c = (idx & 7) << 2;
                #pragma unroll
                for (int j = 0; j < 4; ++j) {
                    unsigned short h, l;
                    splitf(va[i][j], h, l);
                    sA[0][r][c + j] = h;
                    sA[1][r][c + j] = l;
                }
            }
        }
        wait_async0();
        __syncthreads();

        const v16bf aH0 = frag_a(&sA[0][mw * 32 + fr][0], g16);
        const v16bf aL0 = frag_a(&sA[1][mw * 32 + fr][0], g16);
        const v16bf aH1 = frag_a(&sA[0][mw * 32 + 16 + fr][0], g16);
        const v16bf aL1 = frag_a(&sA[1][mw * 32 + 16 + fr][0], g16);

        #pragma unroll
        for (int g = 0; g < 4; ++g) {
            const v16bf bH = frag_b(&sW[g][0][nw * 16 + fr][0], g16);
            const v16bf bL = frag_b(&sW[g][1][nw * 16 + fr][0], g16);
            acc[g][0] = WMMA_BF16(aH0, bH, acc[g][0]);
            acc[g][0] = WMMA_BF16(aH0, bL, acc[g][0]);
            acc[g][0] = WMMA_BF16(aL0, bH, acc[g][0]);
            acc[g][1] = WMMA_BF16(aH1, bH, acc[g][1]);
            acc[g][1] = WMMA_BF16(aH1, bL, acc[g][1]);
            acc[g][1] = WMMA_BF16(aL1, bH, acc[g][1]);
        }
        __syncthreads();
    }

    // Epilogue. C/D layout: N = lane&15, M = elem + 8*(lane>=16)
    const int h = h0 + nw * 16 + fr;
    const float biv = pbi[h], bfv = pbf[h], bov = pbo[h], bgv = pbg[h];

    #pragma unroll
    for (int sub = 0; sub < 2; ++sub) {
        const int mbase = row0 + mw * 32 + sub * 16 + (g16 << 3);
        #pragma unroll
        for (int e = 0; e < 8; ++e) {
            const size_t r   = (size_t)(mbase + e);
            const float iv   = sigmoidf_(acc[0][sub][e] + biv);
            const float fv   = sigmoidf_(acc[1][sub][e] + bfv);
            const float ov   = sigmoidf_(acc[2][sub][e] + bov);
            const float gv   = tanhf(acc[3][sub][e] + bgv);
            const float cp   = cprev[r * HH + h];
            const float cell = fv * cp + iv * gv;
            const float hid  = ov * tanhf(cell);
            outCell[r * HH + h] = cell;
            outHid [r * HH + h] = hid;
        }
    }
}

// ---------------------------------------------------------------------------
// Kernel 2: resid = emb + hidden@Wh.T + bh + ctx@Wc.T + bc   -> d_ws
// ---------------------------------------------------------------------------
__global__ __launch_bounds__(256)
void lstm_resid_bf16x2(const float* __restrict__ emb,
                       const float* __restrict__ hid,
                       const float* __restrict__ ctx,
                       const unsigned short* __restrict__ WhH, const unsigned short* __restrict__ WhL,
                       const unsigned short* __restrict__ WcH, const unsigned short* __restrict__ WcL,
                       const float* __restrict__ pbh, const float* __restrict__ pbc,
                       float* __restrict__ resid)
{
    __shared__ __align__(16) unsigned short sA[2][2][64][SWA];  // [mat][plane][row][K]
    __shared__ __align__(16) unsigned short sW[2][2][64][SWA];  // [mat][plane][e row][K]

    const int tid  = threadIdx.x;
    const int lane = tid & 31;
    const int wave = tid >> 5;
    const int nw   = wave & 3;
    const int mw   = wave >> 2;
    const int row0 = blockIdx.x * 64;
    const int e0   = blockIdx.y * 64;
    const int fr   = lane & 15;
    const int g16  = lane >> 4;

    const float* Asrc[2] = {hid, ctx};
    const unsigned short* WH[2] = {WhH, WcH};
    const unsigned short* WL[2] = {WhL, WcL};

    v8f acc[2] = {};

    for (int kt = 0; kt < HH; kt += 32) {
        // Async stage W planes: 4 combos, 1 chunk each per thread
        {
            const int r = tid >> 2, c = (tid & 3) << 3;
            #pragma unroll
            for (int mp = 0; mp < 4; ++mp) {
                const int m = mp >> 1, p = mp & 1;
                const unsigned short* src =
                    (p ? WL[m] : WH[m]) + (size_t)(e0 + r) * 1024 + kt + c;
                async_load_b128(&sW[m][p][r][c], src);
            }
        }
        // Stage + split both A tiles (loads first, then convert)
        {
            v4f va[2][2];
            #pragma unroll
            for (int m = 0; m < 2; ++m) {
                #pragma unroll
                for (int i = 0; i < 2; ++i) {
                    const int idx = tid + i * 256;
                    const int r = idx >> 3, c = (idx & 7) << 2;
                    va[m][i] = *(const v4f*)(Asrc[m] + (size_t)(row0 + r) * 1024 + kt + c);
                }
            }
            #pragma unroll
            for (int m = 0; m < 2; ++m) {
                #pragma unroll
                for (int i = 0; i < 2; ++i) {
                    const int idx = tid + i * 256;
                    const int r = idx >> 3, c = (idx & 7) << 2;
                    #pragma unroll
                    for (int j = 0; j < 4; ++j) {
                        unsigned short h, l;
                        splitf(va[m][i][j], h, l);
                        sA[m][0][r][c + j] = h;
                        sA[m][1][r][c + j] = l;
                    }
                }
            }
        }
        wait_async0();
        __syncthreads();

        #pragma unroll
        for (int m = 0; m < 2; ++m) {
            const v16bf aH0 = frag_a(&sA[m][0][mw * 32 + fr][0], g16);
            const v16bf aL0 = frag_a(&sA[m][1][mw * 32 + fr][0], g16);
            const v16bf aH1 = frag_a(&sA[m][0][mw * 32 + 16 + fr][0], g16);
            const v16bf aL1 = frag_a(&sA[m][1][mw * 32 + 16 + fr][0], g16);
            const v16bf bH  = frag_b(&sW[m][0][nw * 16 + fr][0], g16);
            const v16bf bL  = frag_b(&sW[m][1][nw * 16 + fr][0], g16);
            acc[0] = WMMA_BF16(aH0, bH, acc[0]);
            acc[0] = WMMA_BF16(aH0, bL, acc[0]);
            acc[0] = WMMA_BF16(aL0, bH, acc[0]);
            acc[1] = WMMA_BF16(aH1, bH, acc[1]);
            acc[1] = WMMA_BF16(aH1, bL, acc[1]);
            acc[1] = WMMA_BF16(aL1, bH, acc[1]);
        }
        __syncthreads();
    }

    const int col = e0 + nw * 16 + fr;
    const float bsum = pbh[col] + pbc[col];
    #pragma unroll
    for (int sub = 0; sub < 2; ++sub) {
        const int mbase = row0 + mw * 32 + sub * 16 + (g16 << 3);
        #pragma unroll
        for (int e = 0; e < 8; ++e) {
            const size_t r = (size_t)(mbase + e);
            resid[r * EE + col] = acc[sub][e] + emb[r * EE + col] + bsum;
        }
    }
}

// ---------------------------------------------------------------------------
// Kernel 3: prediction = resid @ Wp.T + bp   -> d_out[0 : B*E]
// ---------------------------------------------------------------------------
__global__ __launch_bounds__(256)
void lstm_pred_bf16x2(const float* __restrict__ resid,
                      const unsigned short* __restrict__ WpH, const unsigned short* __restrict__ WpL,
                      const float* __restrict__ pbp,
                      float* __restrict__ pred)
{
    __shared__ __align__(16) unsigned short sA[2][64][SWA];
    __shared__ __align__(16) unsigned short sW[2][64][SWA];

    const int tid  = threadIdx.x;
    const int lane = tid & 31;
    const int wave = tid >> 5;
    const int nw   = wave & 3;
    const int mw   = wave >> 2;
    const int row0 = blockIdx.x * 64;
    const int e0   = blockIdx.y * 64;
    const int fr   = lane & 15;
    const int g16  = lane >> 4;

    v8f acc[2] = {};

    for (int kt = 0; kt < EE; kt += 32) {
        {
            const int r = tid >> 2, c = (tid & 3) << 3;
            #pragma unroll
            for (int p = 0; p < 2; ++p) {
                const unsigned short* src =
                    (p ? WpL : WpH) + (size_t)(e0 + r) * 1024 + kt + c;
                async_load_b128(&sW[p][r][c], src);
            }
        }
        {
            v4f va[2];
            #pragma unroll
            for (int i = 0; i < 2; ++i) {
                const int idx = tid + i * 256;
                const int r = idx >> 3, c = (idx & 7) << 2;
                va[i] = *(const v4f*)(resid + (size_t)(row0 + r) * EE + kt + c);
            }
            #pragma unroll
            for (int i = 0; i < 2; ++i) {
                const int idx = tid + i * 256;
                const int r = idx >> 3, c = (idx & 7) << 2;
                #pragma unroll
                for (int j = 0; j < 4; ++j) {
                    unsigned short h, l;
                    splitf(va[i][j], h, l);
                    sA[0][r][c + j] = h;
                    sA[1][r][c + j] = l;
                }
            }
        }
        wait_async0();
        __syncthreads();

        const v16bf aH0 = frag_a(&sA[0][mw * 32 + fr][0], g16);
        const v16bf aL0 = frag_a(&sA[1][mw * 32 + fr][0], g16);
        const v16bf aH1 = frag_a(&sA[0][mw * 32 + 16 + fr][0], g16);
        const v16bf aL1 = frag_a(&sA[1][mw * 32 + 16 + fr][0], g16);
        const v16bf bH  = frag_b(&sW[0][nw * 16 + fr][0], g16);
        const v16bf bL  = frag_b(&sW[1][nw * 16 + fr][0], g16);
        acc[0] = WMMA_BF16(aH0, bH, acc[0]);
        acc[0] = WMMA_BF16(aH0, bL, acc[0]);
        acc[0] = WMMA_BF16(aL0, bH, acc[0]);
        acc[1] = WMMA_BF16(aH1, bH, acc[1]);
        acc[1] = WMMA_BF16(aH1, bL, acc[1]);
        acc[1] = WMMA_BF16(aL1, bH, acc[1]);
        __syncthreads();
    }

    const int col = e0 + nw * 16 + fr;
    const float bv = pbp[col];
    #pragma unroll
    for (int sub = 0; sub < 2; ++sub) {
        const int mbase = row0 + mw * 32 + sub * 16 + (g16 << 3);
        #pragma unroll
        for (int e = 0; e < 8; ++e) {
            const size_t r = (size_t)(mbase + e);
            pred[r * EE + col] = acc[sub][e] + bv;
        }
    }
}

// ---------------------------------------------------------------------------
extern "C" void kernel_launch(void* const* d_in, const int* in_sizes, int n_in,
                              void* d_out, int out_size, void* d_ws, size_t ws_size,
                              hipStream_t stream) {
    const float* emb   = (const float*)d_in[0];
    const float* hprev = (const float*)d_in[1];
    const float* cprev = (const float*)d_in[2];
    const float* ctx   = (const float*)d_in[3];
    const float* Wi = (const float*)d_in[4];   const float* bi = (const float*)d_in[5];
    const float* Wf = (const float*)d_in[6];   const float* bf = (const float*)d_in[7];
    const float* Wo = (const float*)d_in[8];   const float* bo = (const float*)d_in[9];
    const float* Wg = (const float*)d_in[10];  const float* bg = (const float*)d_in[11];
    const float* Wc = (const float*)d_in[12];  const float* bc = (const float*)d_in[13];
    const float* Wh = (const float*)d_in[14];  const float* bh = (const float*)d_in[15];
    const float* Wp = (const float*)d_in[16];  const float* bp = (const float*)d_in[17];

    float* pred = (float*)d_out;                 // [B, E]
    float* hid  = pred + (size_t)BB * EE;        // [B, H]
    float* cell = hid  + (size_t)BB * HH;        // [B, H]

    // Workspace layout: resid fp32, then bf16 hi planes, then bf16 lo planes
    const size_t gateN = (size_t)HH * DD;        // 3,145,728
    const size_t sqN   = (size_t)EE * HH;        // 1,048,576
    const size_t totW  = 4 * gateN + 3 * sqN;    // 15,728,640 elements

    float* resid = (float*)d_ws;
    unsigned short* hiBase = (unsigned short*)((char*)d_ws + (size_t)BB * EE * sizeof(float));
    unsigned short* loBase = hiBase + totW;

    unsigned short* WiH = hiBase;                 unsigned short* WiL = loBase;
    unsigned short* WfH = WiH + gateN;            unsigned short* WfL = WiL + gateN;
    unsigned short* WoH = WfH + gateN;            unsigned short* WoL = WfL + gateN;
    unsigned short* WgH = WoH + gateN;            unsigned short* WgL = WoL + gateN;
    unsigned short* WhH = WgH + gateN;            unsigned short* WhL = WgL + gateN;
    unsigned short* WcH = WhH + sqN;              unsigned short* WcL = WhL + sqN;
    unsigned short* WpH = WcH + sqN;              unsigned short* WpL = WcL + sqN;

    dim3 blk(256);

    // 1) Decompose weights into bf16 hi/lo planes (streaming, ~63MB total)
    {
        const int g4 = (int)(gateN / 4), s4 = (int)(sqN / 4);
        decomp_kernel<<<dim3(g4 / 256), blk, 0, stream>>>(Wi, WiH, WiL, g4);
        decomp_kernel<<<dim3(g4 / 256), blk, 0, stream>>>(Wf, WfH, WfL, g4);
        decomp_kernel<<<dim3(g4 / 256), blk, 0, stream>>>(Wo, WoH, WoL, g4);
        decomp_kernel<<<dim3(g4 / 256), blk, 0, stream>>>(Wg, WgH, WgL, g4);
        decomp_kernel<<<dim3(s4 / 256), blk, 0, stream>>>(Wh, WhH, WhL, s4);
        decomp_kernel<<<dim3(s4 / 256), blk, 0, stream>>>(Wc, WcH, WcL, s4);
        decomp_kernel<<<dim3(s4 / 256), blk, 0, stream>>>(Wp, WpH, WpL, s4);
    }

    // 2) Gates + LSTM elementwise
    lstm_gates_bf16x2<<<dim3(BB / 64, HH / 64), blk, 0, stream>>>(
        emb, hprev, cprev, ctx,
        WiH, WiL, WfH, WfL, WoH, WoL, WgH, WgL,
        bi, bf, bo, bg, hid, cell);

    // 3) Residual
    lstm_resid_bf16x2<<<dim3(BB / 64, EE / 64), blk, 0, stream>>>(
        emb, hid, ctx, WhH, WhL, WcH, WcL, bh, bc, resid);

    // 4) Prediction
    lstm_pred_bf16x2<<<dim3(BB / 64, EE / 64), blk, 0, stream>>>(
        resid, WpH, WpL, bp, pred);
}
